// GAT_18348100288929
// MI455X (gfx1250) — matrix operand order repaired and checked by reading, not compiled, not run on hardware
//
#include <hip/hip_runtime.h>
#include <hip/hip_bf16.h>
#include <stdint.h>

// ---------------------------------------------------------------------------
// GATv2 x2 for MI455X (gfx1250, wave32). Dense GEMMs via v_wmma_f32_16x16x32_f16
// (f16 inputs, f32 accum). Edge softmax via ordered-uint atomicMax + f32 atomicAdd.
// ---------------------------------------------------------------------------

typedef _Float16 v16h __attribute__((ext_vector_type(16)));
typedef float    v8f  __attribute__((ext_vector_type(8)));

#define DC 64   // node channel dim (F == H == 64)
#define EDIM 16 // edge embedding size

// --------------------------------------------------------------------------
// Weight repack: f32 [Din x 64] row-major -> WMMA B-fragment order.
// B (32x16, K x N) layout per ISA 7.12.2: lane l holds column n = l&15;
// VGPR j holds K = (j>>2)*16 + (l>>4)*8 + (j&3)*2 , +1 (within each 32-K block kt).
// Packed index: dword[((kt*4+nt)*32 + lane)*8 + j]. K >= Din rows are zero.
// --------------------------------------------------------------------------
__global__ void pack_weight(const float* __restrict__ W, uint32_t* __restrict__ outp,
                            int Din) {
  int t    = threadIdx.x;       // 0..255  => (kt, nt, lane)
  int lane = t & 31;
  int nt   = (t >> 5) & 3;
  int kt   = t >> 7;
  int n0   = lane & 15;
  int khi  = lane >> 4;
  int n    = nt * 16 + n0;
#pragma unroll
  for (int j = 0; j < 8; ++j) {
    int k = kt * 32 + (j >> 2) * 16 + khi * 8 + (j & 3) * 2;
    union { _Float16 h[2]; uint32_t u; } p;
    p.h[0] = (k     < Din) ? (_Float16)W[(size_t)k       * DC + n] : (_Float16)0.0f;
    p.h[1] = (k + 1 < Din) ? (_Float16)W[(size_t)(k + 1) * DC + n] : (_Float16)0.0f;
    outp[((size_t)(kt * 4 + nt) * 32 + lane) * 8 + j] = p.u;
  }
}

// convert two float4 (8 consecutive K values) into halves [base..base+7] of a frag
__device__ __forceinline__ void cvt8(v16h& a, int base, float4 x, float4 y) {
  a[base + 0] = (_Float16)x.x; a[base + 1] = (_Float16)x.y;
  a[base + 2] = (_Float16)x.z; a[base + 3] = (_Float16)x.w;
  a[base + 4] = (_Float16)y.x; a[base + 5] = (_Float16)y.y;
  a[base + 6] = (_Float16)y.z; a[base + 7] = (_Float16)y.w;
}

// one 16x64 GEMM tile (f16 WMMA, K=64) + bias -> out[16][64]
// Hot path: full tile (row0+16 <= nrows) -> unguarded stores, immediate offsets.
__device__ __forceinline__ void mat16x64(v16h a0, v16h a1,
                                         const uint32_t* __restrict__ wp,
                                         const float* __restrict__ bias,
                                         float* __restrict__ out, int lane,
                                         long row0, int nrows) {
  int n0 = lane & 15, half = lane >> 4;
  v8f d[4];
#pragma unroll
  for (int nt = 0; nt < 4; ++nt) {
    v8f c = {};
    v16h b0 = ((const v16h*)wp)[nt * 32 + lane];
    c = __builtin_amdgcn_wmma_f32_16x16x32_f16(false, a0, false, b0, (short)0, c,
                                               false, false);
    v16h b1 = ((const v16h*)wp)[(4 + nt) * 32 + lane];
    d[nt] = __builtin_amdgcn_wmma_f32_16x16x32_f16(false, a1, false, b1, (short)0, c,
                                                   false, false);
  }
  float* orow = out + (size_t)(8 * half) * DC + n0;   // single base, imm offsets below
  if (row0 + 16 <= (long)nrows) {                     // uniform per-wave fast path
#pragma unroll
    for (int nt = 0; nt < 4; ++nt) {
      float bb = bias[nt * 16 + n0];
#pragma unroll
      for (int r = 0; r < 8; ++r)
        orow[(size_t)r * DC + nt * 16] = d[nt][r] + bb;
    }
  } else {                                            // rare ragged tail
#pragma unroll
    for (int nt = 0; nt < 4; ++nt) {
      float bb = bias[nt * 16 + n0];
#pragma unroll
      for (int r = 0; r < 8; ++r)
        if (row0 + r + 8 * half < (long)nrows)
          orow[(size_t)r * DC + nt * 16] = d[nt][r] + bb;
    }
  }
}

// --------------------------------------------------------------------------
// Node transforms: xl = xWl+bl, xr = xWr+br, acc = xWres+bres (residual seed)
// One wave computes a 16-row tile of all three outputs.
// A (16x32 f16) layout per ISA 7.12.2: lane l -> row m = l&15; VGPR j holds
// K = (j>>2)*16 + (l>>4)*8 + (j&3)*2 , +1  => two contiguous 8-float runs.
// --------------------------------------------------------------------------
__global__ void node_gemm(const float* __restrict__ x,
                          const uint32_t* __restrict__ wl,
                          const uint32_t* __restrict__ wr,
                          const uint32_t* __restrict__ wres,
                          const float* __restrict__ bl,
                          const float* __restrict__ br,
                          const float* __restrict__ bres,
                          float* __restrict__ xl, float* __restrict__ xr,
                          float* __restrict__ acc, int nrows) {
  int lane  = threadIdx.x & 31;
  long tile = (long)blockIdx.x * (blockDim.x >> 5) + (threadIdx.x >> 5);
  long row0 = tile * 16;
  if (row0 >= nrows) return;

  int m = lane & 15, half = lane >> 4;
  long rr = row0 + m;
  if (rr >= nrows) rr = nrows - 1;
  const float* xp = x + (size_t)rr * DC + half * 8;

  v16h a0, a1;
  cvt8(a0, 0, *(const float4*)(xp +  0), *(const float4*)(xp +  4)); // K 0..7 (+h*8)
  cvt8(a0, 8, *(const float4*)(xp + 16), *(const float4*)(xp + 20)); // K 16..23
  cvt8(a1, 0, *(const float4*)(xp + 32), *(const float4*)(xp + 36)); // K 32..39
  cvt8(a1, 8, *(const float4*)(xp + 48), *(const float4*)(xp + 52)); // K 48..55

  size_t ob = (size_t)row0 * DC;
  mat16x64(a0, a1, wl,   bl,   xl  + ob, lane, row0, nrows);
  mat16x64(a0, a1, wr,   br,   xr  + ob, lane, row0, nrows);
  mat16x64(a0, a1, wres, bres, acc + ob, lane, row0, nrows);
}

__device__ __forceinline__ unsigned ordered_u32(float f) {
  unsigned u = __float_as_uint(f);
  return (u & 0x80000000u) ? ~u : (u | 0x80000000u);
}
__device__ __forceinline__ float unordered_f32(unsigned u) {
  return __uint_as_float((u & 0x80000000u) ? (u ^ 0x80000000u) : ~u);
}

// --------------------------------------------------------------------------
// Edge logits: s[e] = leaky_relu(xl[src] + xr[dst] + ea@We) . a
// One wave per 16-edge tile. ea@We via one WMMA per 16-col tile (K padded 16->32,
// both the A fragment tail and the packed We K>=16 rows are zero).
// --------------------------------------------------------------------------
__global__ void edge_logits(const float* __restrict__ ea,
                            const int* __restrict__ srcI,
                            const int* __restrict__ dstI,
                            const uint32_t* __restrict__ wep,
                            const float* __restrict__ xl,
                            const float* __restrict__ xr,
                            const float* __restrict__ avec,
                            float* __restrict__ sbuf,
                            unsigned* __restrict__ smax, long nedges) {
  int lane  = threadIdx.x & 31;
  long tile = (long)blockIdx.x * (blockDim.x >> 5) + (threadIdx.x >> 5);
  long e0   = tile * 16;
  if (e0 >= nedges) return;

  int m = lane & 15, half = lane >> 4, n0 = m;
  long er = e0 + m; if (er >= nedges) er = nedges - 1;
  const float* arow = ea + (size_t)er * EDIM + half * 8;  // K = half*8 + 0..7

  v16h a;
  cvt8(a, 0, *(const float4*)(arow), *(const float4*)(arow + 4));
#pragma unroll
  for (int j = 8; j < 16; ++j) a[j] = (_Float16)0.0f;     // K >= 16 padding

  v8f d[4];
#pragma unroll
  for (int nt = 0; nt < 4; ++nt) {
    v8f c = {};
    v16h b = ((const v16h*)wep)[nt * 32 + lane];          // kt = 0 fragments only
    d[nt] = __builtin_amdgcn_wmma_f32_16x16x32_f16(false, a, false, b, (short)0, c,
                                                   false, false);
  }

  float partial[8];
#pragma unroll
  for (int r = 0; r < 8; ++r) {
    long e = e0 + r + 8 * half;
    long ec = (e < nedges) ? e : (nedges - 1);
    long sn = (long)srcI[ec];
    long dn = (long)dstI[ec];
    const float* xls = xl + (size_t)sn * DC;
    const float* xrd = xr + (size_t)dn * DC;
    float accv = 0.0f;
#pragma unroll
    for (int nt = 0; nt < 4; ++nt) {
      int c0  = nt * 16 + n0;
      float v = d[nt][r] + xls[c0] + xrd[c0];
      v = (v > 0.0f) ? v : 0.2f * v;           // leaky_relu 0.2
      accv += v * avec[c0];
    }
    partial[r] = accv;
  }
  // reduce each row over its 16-lane half-group
#pragma unroll
  for (int r = 0; r < 8; ++r) {
    partial[r] += __shfl_xor(partial[r], 1);
    partial[r] += __shfl_xor(partial[r], 2);
    partial[r] += __shfl_xor(partial[r], 4);
    partial[r] += __shfl_xor(partial[r], 8);
  }
  if (n0 < 8) {                                // lanes 0-7 / 16-23 write 8 edges each
    long e = e0 + n0 + 8 * half;
    if (e < nedges) {
      float sv = partial[0];
#pragma unroll
      for (int r = 1; r < 8; ++r)
        if (n0 == r) sv = partial[r];
      sbuf[e] = sv;
      atomicMax(&smax[(long)dstI[e]], ordered_u32(sv));
    }
  }
}

__global__ void edge_softmax(const float* __restrict__ sbuf,
                             const int* __restrict__ dstI,
                             const unsigned* __restrict__ smax,
                             float* __restrict__ exbuf, float* __restrict__ den,
                             long nedges) {
  long e = (long)blockIdx.x * blockDim.x + threadIdx.x;
  if (e >= nedges) return;
  long dn   = (long)dstI[e];
  float mx  = unordered_f32(smax[dn]);
  float exv = __expf(sbuf[e] - mx);
  exbuf[e]  = exv;
  atomicAdd(&den[dn], exv);
}

// out[dst] += alpha * xl[src]; 16 lanes per edge, 4 channels each (float4)
__global__ void edge_aggregate(const float* __restrict__ exbuf,
                               const float* __restrict__ den,
                               const int* __restrict__ srcI,
                               const int* __restrict__ dstI,
                               const float* __restrict__ xl,
                               float* __restrict__ acc, long nedges) {
  long t = (long)blockIdx.x * blockDim.x + threadIdx.x;
  long e = t >> 4;
  if (e >= nedges) return;
  int  j  = (int)(t & 15);
  long dn = (long)dstI[e];
  long sn = (long)srcI[e];
  float alpha = exbuf[e] / (den[dn] + 1e-16f);
  float4 v = ((const float4*)xl)[(size_t)sn * 16 + j];
  float* o = acc + (size_t)dn * DC + j * 4;
  atomicAdd(o + 0, alpha * v.x);
  atomicAdd(o + 1, alpha * v.y);
  atomicAdd(o + 2, alpha * v.z);
  atomicAdd(o + 3, alpha * v.w);
}

__global__ void bias_act(const float* __restrict__ in, const float* __restrict__ cb,
                         float* __restrict__ out, long n, int do_relu) {
  long i = (long)blockIdx.x * blockDim.x + threadIdx.x;
  if (i >= n) return;
  float v = in[i] + cb[i & (DC - 1)];
  if (do_relu) v = (v > 0.0f) ? v : 0.0f;
  out[i] = v;
}

__global__ void init_seg(unsigned* __restrict__ smax, float* __restrict__ den, int n) {
  int i = blockIdx.x * blockDim.x + threadIdx.x;
  if (i < n) { smax[i] = 0x007FFFFFu; /* ordered(-inf) */ den[i] = 0.0f; }
}

// ---------------------------------------------------------------------------
extern "C" void kernel_launch(void* const* d_in, const int* in_sizes, int n_in,
                              void* d_out, int out_size, void* d_ws, size_t ws_size,
                              hipStream_t stream) {
  const float* x    = (const float*)d_in[0];
  const int*   ei   = (const int*)d_in[1];      // edge_index [2, E]
  const float* ea   = (const float*)d_in[2];
  const float* Wl1  = (const float*)d_in[3];
  const float* bl1  = (const float*)d_in[4];
  const float* Wr1  = (const float*)d_in[5];
  const float* br1  = (const float*)d_in[6];
  const float* We1  = (const float*)d_in[7];
  const float* a1   = (const float*)d_in[8];
  const float* cb1  = (const float*)d_in[9];
  const float* L1w  = (const float*)d_in[10];
  const float* L1b  = (const float*)d_in[11];
  const float* Wl2  = (const float*)d_in[12];
  const float* bl2  = (const float*)d_in[13];
  const float* Wr2  = (const float*)d_in[14];
  const float* br2  = (const float*)d_in[15];
  const float* We2  = (const float*)d_in[16];
  const float* a2   = (const float*)d_in[17];
  const float* cb2  = (const float*)d_in[18];
  const float* L2w  = (const float*)d_in[19];
  const float* L2b  = (const float*)d_in[20];

  const int  Nn = in_sizes[0] / DC;             // 100000
  const long Ee = (long)in_sizes[2] / EDIM;     // 1600000
  const int* srcI = ei;
  const int* dstI = ei + Ee;

  float* out = (float*)d_out;

  // ---- workspace layout (256B aligned chunks) ----
  uint8_t* w = (uint8_t*)d_ws;
  uint32_t* packs = (uint32_t*)w;               // 8 mats x 8192 B
  size_t off = 8 * 8192;
  auto alignup = [](size_t v) { return (v + 255) & ~(size_t)255; };
  float* xl   = (float*)(w + off); off += alignup((size_t)Nn * DC * 4);
  float* xr   = (float*)(w + off); off += alignup((size_t)Nn * DC * 4);
  float* hacc = (float*)(w + off); off += alignup((size_t)Nn * DC * 4);
  float* sbuf = (float*)(w + off); off += alignup((size_t)Ee * 4);
  float* exbf = (float*)(w + off); off += alignup((size_t)Ee * 4);
  unsigned* smax = (unsigned*)(w + off); off += alignup((size_t)Nn * 4);
  float*    den  = (float*)(w + off);    off += alignup((size_t)Nn * 4);
  (void)ws_size; (void)n_in; (void)out_size;

  uint32_t* pWl1 = packs + 0 * 2048;
  uint32_t* pWr1 = packs + 1 * 2048;
  uint32_t* pL1  = packs + 2 * 2048;
  uint32_t* pWl2 = packs + 3 * 2048;
  uint32_t* pWr2 = packs + 4 * 2048;
  uint32_t* pL2  = packs + 5 * 2048;
  uint32_t* pWe1 = packs + 6 * 2048;
  uint32_t* pWe2 = packs + 7 * 2048;

  // ---- weight repack (tiny, once per call) ----
  pack_weight<<<1, 256, 0, stream>>>(Wl1, pWl1, DC);
  pack_weight<<<1, 256, 0, stream>>>(Wr1, pWr1, DC);
  pack_weight<<<1, 256, 0, stream>>>(L1w, pL1, DC);
  pack_weight<<<1, 256, 0, stream>>>(Wl2, pWl2, DC);
  pack_weight<<<1, 256, 0, stream>>>(Wr2, pWr2, DC);
  pack_weight<<<1, 256, 0, stream>>>(L2w, pL2, DC);
  pack_weight<<<1, 256, 0, stream>>>(We1, pWe1, EDIM);
  pack_weight<<<1, 256, 0, stream>>>(We2, pWe2, EDIM);

  const int  waveTilesN = (Nn + 15) / 16;
  const int  gemmGrid   = (waveTilesN + 3) / 4;           // 4 waves / block
  const long edgeTiles  = (Ee + 15) / 16;
  const int  edgeGrid   = (int)((edgeTiles + 3) / 4);
  const int  initGrid   = (Nn + 255) / 256;
  const int  eGrid      = (int)((Ee + 255) / 256);
  const int  aggGrid    = (int)((Ee * 16 + 255) / 256);
  const long nElems     = (long)Nn * DC;
  const int  pwGrid     = (int)((nElems + 255) / 256);

  // =================== Layer 1 ===================
  node_gemm<<<gemmGrid, 128, 0, stream>>>(x, pWl1, pWr1, pL1, bl1, br1, L1b,
                                          xl, xr, hacc, Nn);
  init_seg<<<initGrid, 256, 0, stream>>>(smax, den, Nn);
  edge_logits<<<edgeGrid, 128, 0, stream>>>(ea, srcI, dstI, pWe1, xl, xr, a1,
                                            sbuf, smax, Ee);
  edge_softmax<<<eGrid, 256, 0, stream>>>(sbuf, dstI, smax, exbf, den, Ee);
  edge_aggregate<<<aggGrid, 256, 0, stream>>>(exbf, den, srcI, dstI, xl, hacc, Ee);
  bias_act<<<pwGrid, 256, 0, stream>>>(hacc, cb1, hacc, nElems, 1);  // h = relu(.)

  // =================== Layer 2 ===================
  node_gemm<<<gemmGrid, 128, 0, stream>>>(hacc, pWl2, pWr2, pL2, bl2, br2, L2b,
                                          xl, xr, out, Nn);
  init_seg<<<initGrid, 256, 0, stream>>>(smax, den, Nn);
  edge_logits<<<edgeGrid, 128, 0, stream>>>(ea, srcI, dstI, pWe2, xl, xr, a2,
                                            sbuf, smax, Ee);
  edge_softmax<<<eGrid, 256, 0, stream>>>(sbuf, dstI, smax, exbf, den, Ee);
  edge_aggregate<<<aggGrid, 256, 0, stream>>>(exbf, den, srcI, dstI, xl, out, Ee);
  bias_act<<<pwGrid, 256, 0, stream>>>(out, cb2, out, nElems, 0);
}
